// xLSTM_27668179320997
// MI455X (gfx1250) — matrix operand order repaired and checked by reading, not compiled, not run on hardware
//
#include <hip/hip_runtime.h>

// ---------------------------------------------------------------------------
// xLSTM (2 recurrent layers, 5 gates) + FC for MI455X (gfx1250, wave32).
// Per-time-step work is ONE fused kernel: bf16 WMMA gate GEMM (split-K)
// -> LDS -> gate nonlinearity -> state update. Weights pre-transposed to
// [N][K] bf16 so A and B fragments are contiguous 16B/32B global loads.
// Wave coordinates go through readfirstlane so segment selection is a
// scalar branch (no per-iteration v_cndmask pointer selects, EXEC all-1s).
// ---------------------------------------------------------------------------

typedef __attribute__((ext_vector_type(16))) __bf16 v16bf;
typedef __attribute__((ext_vector_type(8)))  float  v8f;

union Frag {
    v16bf v;
    uint4 u[2];
};

__device__ __forceinline__ unsigned short f2bf(float f) {
    unsigned int u = __float_as_uint(f);
    unsigned int r = u + 0x7FFFu + ((u >> 16) & 1u);   // round to nearest even
    return (unsigned short)(r >> 16);
}

// A fragment: 16x32 bf16 (M x K). lanes 0-15 => M rows; halves 0-7 = K[0..7]
// (lanes 16-31: K[8..15]); halves 8-15 = K[16..23] (lanes 16-31: K[24..31]).
__device__ __forceinline__ void load_a_frag(Frag& f, const unsigned short* __restrict__ A,
                                            int lda, int row, int k0, int laneHi) {
    const unsigned short* p = A + (size_t)row * lda + k0 + laneHi * 8;
    f.u[0] = *(const uint4*)p;
    f.u[1] = *(const uint4*)(p + 16);
}

// B fragment: 32x16 bf16 (K x N) from transposed weight Bt[N][K]. lanes 0-15
// hold K=0..7,16..23; lanes 16-31 hold K=8..15,24..31 (same as A).
__device__ __forceinline__ void load_b_frag(Frag& f, const unsigned short* __restrict__ Bt,
                                            int ldb, int nrow, int k0, int laneHi) {
    const unsigned short* p = Bt + (size_t)nrow * ldb + k0 + laneHi * 8;
    f.u[0] = *(const uint4*)p;
    f.u[1] = *(const uint4*)(p + 16);
}

__device__ __forceinline__ float sigmoidf_(float x) {
    return 1.f / (1.f + __expf(-x));
}

// One K-segment of WMMA accumulation with compile-time bounds.
// Global k in [KB, KE); A is indexed at (k - AOFF). Empty if KB >= KE.
template <int KB, int KE, int AOFF, int LDB>
__device__ __forceinline__ void mma_seg(v8f& acc,
    const unsigned short* __restrict__ A, int lda,
    const unsigned short* __restrict__ Bt,
    int arow, int brow, int laneHi)
{
#pragma unroll 4
    for (int k = KB; k < KE; k += 32) {
        Frag a, b;
        load_a_frag(a, A, lda, arow, k - AOFF, laneHi);
        load_b_frag(b, Bt, LDB, brow, k, laneHi);
        acc = __builtin_amdgcn_wmma_f32_16x16x32_bf16(
            false, a.v, false, b.v, (short)0, acc, false, false);
    }
}

// ---------------------------------------------------------------------------
// Fused xLSTM step: for one time step t,
//   g[b, gate*H + hc] = sum_k [h | x_t][b,k] * Wcat_t[gate*H+hc, k]
// then gate math + state update, all in one kernel.
// Block: 640 threads = 20 waves = waveM(2) x gate(5) x kHalf(2).
// Each block owns a 16-column slice of H (all 5 gates, all 32 batch rows).
// Grid: H/16 = 64 blocks.
// ---------------------------------------------------------------------------
template <int K1, int K2>
__global__ __launch_bounds__(640) void lstm_step_fused(
    const unsigned short* __restrict__ A1, int lda1,   // h state  [32, K1] bf16
    const unsigned short* __restrict__ A2, int lda2,   // x_t rows [32, K2] bf16
    const unsigned short* __restrict__ Bt,             // Wcat_t [5H, K1+K2] bf16
    const float* __restrict__ bias,                    // [5H]
    float* __restrict__ c,                             // [32, H] f32 state
    unsigned short* __restrict__ h,                    // [32, H] bf16 state
    unsigned short* __restrict__ hseq,                 // [B, T, H] bf16 out
    int Hd, int T, int t)
{
    constexpr int KT = K1 + K2;
    constexpr int KH = KT / 2;
    constexpr int S0 = (KH < K1) ? KH : K1;   // seg1 end within first half
    constexpr int S1 = (K1 > KH) ? K1 : KH;   // seg1/seg2 boundary in 2nd half

    __shared__ float lds_g[2][5][32][16];              // [kHalf][gate][m][col]

    const int tid    = threadIdx.x;
    const int lane   = tid & 31;
    // Wave coordinates as SGPRs -> scalar branches, scalar tile bases.
    const int wave   = __builtin_amdgcn_readfirstlane(tid >> 5);   // 0..19
    const int kHalf  = wave & 1;
    const int gate   = (wave >> 1) % 5;
    const int waveM  = (wave >> 1) / 5;                // 0..1
    const int laneHi = lane >> 4;
    const int mr     = lane & 15;

    const int hc0  = blockIdx.x * 16;
    const int arow = waveM * 16 + mr;                  // batch row
    const int brow = gate * Hd + hc0 + mr;             // gate weight row

    v8f acc = {0.f, 0.f, 0.f, 0.f, 0.f, 0.f, 0.f, 0.f};

    if (kHalf == 0) {
        mma_seg<0,  S0, 0,  KT>(acc, A1, lda1, Bt, arow, brow, laneHi);
        mma_seg<S0, KH, K1, KT>(acc, A2, lda2, Bt, arow, brow, laneHi);
    } else {
        mma_seg<KH, S1, 0,  KT>(acc, A1, lda1, Bt, arow, brow, laneHi);
        mma_seg<S1, KT, K1, KT>(acc, A2, lda2, Bt, arow, brow, laneHi);
    }

#pragma unroll
    for (int r = 0; r < 8; ++r)
        lds_g[kHalf][gate][waveM * 16 + laneHi * 8 + r][mr] = acc[r];
    __syncthreads();

    // 512 elements (32 batch x 16 cols); threads 0..511 handle one each.
    if (tid < 512) {
        const int m   = tid >> 4;                      // batch row
        const int col = tid & 15;
        const int hc  = hc0 + col;

        float gi = lds_g[0][0][m][col] + lds_g[1][0][m][col] + bias[0 * Hd + hc];
        float gf = lds_g[0][1][m][col] + lds_g[1][1][m][col] + bias[1 * Hd + hc];
        float gc = lds_g[0][2][m][col] + lds_g[1][2][m][col] + bias[2 * Hd + hc];
        float go = lds_g[0][3][m][col] + lds_g[1][3][m][col] + bias[3 * Hd + hc];
        float gm = lds_g[0][4][m][col] + lds_g[1][4][m][col] + bias[4 * Hd + hc];

        float i  = sigmoidf_(gi);
        float f  = sigmoidf_(gf);
        float ct = tanhf(gc);
        float o  = sigmoidf_(go);
        float mn = tanhf(gm);

        const int sidx = m * Hd + hc;
        float cn = f * c[sidx] + i * ct;
        c[sidx] = cn;
        float hn = o * tanhf(cn) + mn;

        unsigned short hb = f2bf(hn);
        h[sidx] = hb;
        hseq[(size_t)m * T * Hd + (size_t)t * Hd + hc] = hb;
    }
}

// ---------------------------------------------------------------------------
// Plain bf16 WMMA GEMM (used for the final FC).
// C[M,N] f32 = A[M,K] (bf16) x Bt[N,K]^T (bf16) + bias[N].
// Block: 256 threads = 8 waves arranged 2(M) x 4(N); block tile 32 x 64.
// ---------------------------------------------------------------------------
template <int K>
__global__ __launch_bounds__(256) void gemm_bf16_wmma(
    const unsigned short* __restrict__ A, int lda,
    const unsigned short* __restrict__ Bt,
    float* __restrict__ C, int ldc,
    const float* __restrict__ bias)
{
    const int lane   = threadIdx.x & 31;
    const int wave   = __builtin_amdgcn_readfirstlane(threadIdx.x >> 5);
    const int waveM  = wave >> 2;
    const int waveN  = wave & 3;
    const int m0     = blockIdx.y * 32 + waveM * 16;
    const int n0     = blockIdx.x * 64 + waveN * 16;
    const int laneHi = lane >> 4;
    const int mr     = lane & 15;

    v8f acc = {0.f, 0.f, 0.f, 0.f, 0.f, 0.f, 0.f, 0.f};
    const int arow = m0 + mr;
    const int brow = n0 + mr;

    mma_seg<0, K, 0, K>(acc, A, lda, Bt, arow, brow, laneHi);

    float bn = bias ? bias[n0 + mr] : 0.f;
    float* cp = C + (size_t)(m0 + laneHi * 8) * ldc + n0 + mr;
#pragma unroll
    for (int r = 0; r < 8; ++r)
        cp[(size_t)r * ldc] = acc[r] + bn;
}

// Zero-init h (bf16) and c (f32) states.
__global__ void init_state(unsigned short* __restrict__ h, float* __restrict__ c, int n)
{
    int idx = blockIdx.x * blockDim.x + threadIdx.x;
    if (idx < n) { h[idx] = 0; c[idx] = 0.f; }
}

// f32 -> bf16 elementwise cast.
__global__ void cast_bf16(const float* __restrict__ in, unsigned short* __restrict__ out, int n)
{
    int idx = blockIdx.x * blockDim.x + threadIdx.x;
    if (idx < n) out[idx] = f2bf(in[idx]);
}

// in: f32 [K][N] row-major -> out: bf16 [N][ldOut] (transposed).
__global__ void transpose_cast(const float* __restrict__ in, unsigned short* __restrict__ out,
                               int K, int N, int ldOut)
{
    int idx = blockIdx.x * blockDim.x + threadIdx.x;
    if (idx >= K * N) return;
    int k = idx / N;
    int n = idx - k * N;
    out[(size_t)n * ldOut + k] = f2bf(in[idx]);
}

// ---------------------------------------------------------------------------

extern "C" void kernel_launch(void* const* d_in, const int* in_sizes, int n_in,
                              void* d_out, int out_size, void* d_ws, size_t ws_size,
                              hipStream_t stream) {
    (void)in_sizes; (void)n_in; (void)out_size; (void)ws_size;

    constexpr int B = 32, T = 512, IN = 512, H = 1024, OUT = 512;
    constexpr int H5 = 5 * H;                 // 5120
    constexpr int KC0 = H + IN;               // 1536  (layer0: [h | x_t])
    constexpr int KC1 = H + H;                // 2048  (layer1: [h | h1_t])

    const float* x   = (const float*)d_in[0];
    const float* Wx0 = (const float*)d_in[1];
    const float* Wh0 = (const float*)d_in[2];
    const float* b0  = (const float*)d_in[3];
    const float* Wx1 = (const float*)d_in[4];
    const float* Wh1 = (const float*)d_in[5];
    const float* b1  = (const float*)d_in[6];
    const float* Wfc = (const float*)d_in[7];
    const float* bfc = (const float*)d_in[8];
    float* out = (float*)d_out;

    // --- workspace carve-up (256B aligned) ---
    char* ws = (char*)d_ws;
    size_t off = 0;
    auto alloc = [&](size_t bytes) {
        size_t o = off;
        off = (off + bytes + 255) & ~(size_t)255;
        return (void*)(ws + o);
    };
    unsigned short* xbf   = (unsigned short*)alloc((size_t)B * T * IN * 2);
    unsigned short* Wc0   = (unsigned short*)alloc((size_t)H5 * KC0 * 2);   // [5H][H+IN]
    unsigned short* Wc1   = (unsigned short*)alloc((size_t)H5 * KC1 * 2);   // [5H][2H]
    unsigned short* Wfct  = (unsigned short*)alloc((size_t)OUT * H * 2);    // [OUT][H]
    unsigned short* h1seq = (unsigned short*)alloc((size_t)B * T * H * 2);
    unsigned short* h2seq = (unsigned short*)alloc((size_t)B * T * H * 2);
    unsigned short* hst0  = (unsigned short*)alloc((size_t)B * H * 2);
    unsigned short* hst1  = (unsigned short*)alloc((size_t)B * H * 2);
    float* cst = (float*)alloc((size_t)B * H * 4);

    const int TPB = 256;
    auto blocks = [](size_t n, int tpb) { return (unsigned)((n + tpb - 1) / tpb); };

    // --- one-time prep: casts + weight transposes (concat [Wh | Wx] along K) ---
    cast_bf16<<<blocks((size_t)B * T * IN, TPB), TPB, 0, stream>>>(x, xbf, B * T * IN);
    transpose_cast<<<blocks((size_t)H * H5, TPB), TPB, 0, stream>>>(Wh0, Wc0,      H,  H5, KC0);
    transpose_cast<<<blocks((size_t)IN * H5, TPB), TPB, 0, stream>>>(Wx0, Wc0 + H, IN, H5, KC0);
    transpose_cast<<<blocks((size_t)H * H5, TPB), TPB, 0, stream>>>(Wh1, Wc1,      H,  H5, KC1);
    transpose_cast<<<blocks((size_t)H * H5, TPB), TPB, 0, stream>>>(Wx1, Wc1 + H,  H,  H5, KC1);
    transpose_cast<<<blocks((size_t)H * OUT, TPB), TPB, 0, stream>>>(Wfc, Wfct,    H,  OUT, H);

    const dim3 stepGrid(H / 16);              // 64 blocks, each owns 16 h-cols
    const dim3 stepBlk(640);                  // 20 waves: 2(M) x 5(gate) x 2(splitK)
    unsigned initBlocks = blocks((size_t)B * H, TPB);

    // --- layer 0: fused  g = [h | x_t] @ [Wh0 ; Wx0]^T  + gates ---
    init_state<<<initBlocks, TPB, 0, stream>>>(hst0, cst, B * H);
    for (int t = 0; t < T; ++t) {
        lstm_step_fused<H, IN><<<stepGrid, stepBlk, 0, stream>>>(
            hst0, H, xbf + (size_t)t * IN, T * IN, Wc0, b0,
            cst, hst0, h1seq, H, T, t);
    }

    // --- layer 1: fused  g = [h | h1_t] @ [Wh1 ; Wx1]^T  + gates ---
    init_state<<<initBlocks, TPB, 0, stream>>>(hst1, cst, B * H);
    for (int t = 0; t < T; ++t) {
        lstm_step_fused<H, H><<<stepGrid, stepBlk, 0, stream>>>(
            hst1, H, h1seq + (size_t)t * H, T * H, Wc1, b1,
            cst, hst1, h2seq, H, T, t);
    }

    // --- FC: out[B*T, OUT] = h2seq @ Wfc + bfc ---
    const dim3 fcGrid(OUT / 64, (B * T) / 32);  // (8, 512)
    gemm_bf16_wmma<H><<<fcGrid, dim3(256), 0, stream>>>(
        h2seq, H, Wfct, out, OUT, bfc);
}
